// CogAgentDecoderLayer_1692217115223
// MI455X (gfx1250) — matrix operand order repaired and checked by reading, not compiled, not run on hardware
//
#include <hip/hip_runtime.h>
#include <cstdint>
#include <cstddef>

// ---- problem constants (match reference) ----
#define S_  2048
#define H_  2048
#define E_  2048
#define CH_ 1024
#define CC_ 1024
#define NH_ 16
#define HD_ 128
#define CHD_ 64
#define I_  5504

typedef __attribute__((ext_vector_type(16))) __bf16   bf16x16;
typedef __attribute__((ext_vector_type(8)))  float    floatx8;
typedef __attribute__((ext_vector_type(4)))  uint32_t uintx4;
typedef __attribute__((ext_vector_type(4)))  int      int4v;

union ABFrag { bf16x16 v; uint32_t u[8]; };

__device__ __forceinline__ __bf16 f2bf(float f) {
  uint32_t u = __builtin_bit_cast(uint32_t, f);
  uint32_t r = u + 0x7FFFu + ((u >> 16) & 1u);   // round-to-nearest-even
  uint16_t h = (uint16_t)(r >> 16);
  return __builtin_bit_cast(__bf16, h);
}

__device__ __forceinline__ floatx8 wmma_bf16(const ABFrag& a, const ABFrag& b, floatx8 c) {
  // D(f32,16x16) = A(bf16,16x32) x B(bf16,32x16) + C
  return __builtin_amdgcn_wmma_f32_16x16x32_bf16(false, a.v, false, b.v, (short)0, c, false, false);
}

// ---- CDNA5 async global->LDS copy (guarded; falls back to plain copy) ----
#if defined(__has_builtin)
#if __has_builtin(__builtin_amdgcn_global_load_async_to_lds_b128) && \
    __has_builtin(__builtin_amdgcn_s_wait_asynccnt)
#define USE_ASYNC_LDS 1
#endif
#endif

#ifdef USE_ASYNC_LDS
// param types per clang: (v4i __device__* src, v4i __shared__* dst, imm offset, imm cpol)
typedef __attribute__((address_space(1))) int4v* gas4_t;
typedef __attribute__((address_space(3))) int4v* las4_t;
#endif

__device__ __forceinline__ void stage16(const __bf16* g, __bf16* l) {
#ifdef USE_ASYNC_LDS
  __builtin_amdgcn_global_load_async_to_lds_b128(
      (gas4_t)(uintptr_t)g, (las4_t)(uintptr_t)l, 0, 0);
#else
  *(uintx4*)l = *(const uintx4*)g;
#endif
}
__device__ __forceinline__ void stage_wait() {
#ifdef USE_ASYNC_LDS
  __builtin_amdgcn_s_wait_asynccnt(0);
#endif
}

// =====================================================================
// Elementwise kernels
// =====================================================================
__global__ void cvt_f32_bf16(const float* __restrict__ src, __bf16* __restrict__ dst, size_t n) {
  for (size_t i = (size_t)blockIdx.x * blockDim.x + threadIdx.x; i < n;
       i += (size_t)gridDim.x * blockDim.x)
    dst[i] = f2bf(src[i]);
}

// fp32 W[K][N] -> bf16 Wt[N][K] (tiled transpose). K,N % 32 == 0.
__global__ void cvt_transpose(const float* __restrict__ src, __bf16* __restrict__ dst,
                              int K, int N) {
  __shared__ __bf16 tile[32][33];
  int n0 = blockIdx.x * 32, k0 = blockIdx.y * 32;
  int tx = threadIdx.x;
  for (int r = threadIdx.y; r < 32; r += 8)
    tile[r][tx] = f2bf(src[(size_t)(k0 + r) * N + n0 + tx]);
  __syncthreads();
  for (int r = threadIdx.y; r < 32; r += 8)
    dst[(size_t)(n0 + r) * K + k0 + tx] = tile[tx][r];
}

// rmsnorm one row per block; optional vision mask split into two bf16 outputs.
__global__ void rmsnorm_mask(const float* __restrict__ x, const float* __restrict__ w,
                             const unsigned char* __restrict__ vmask,
                             __bf16* __restrict__ out_v, __bf16* __restrict__ out_l, int H) {
  int r = blockIdx.x, t = threadIdx.x;
  const float* row = x + (size_t)r * H;
  float ss = 0.f;
  for (int i = t; i < H; i += 256) { float v = row[i]; ss += v * v; }
  __shared__ float red[256];
  red[t] = ss; __syncthreads();
  for (int o = 128; o > 0; o >>= 1) { if (t < o) red[t] += red[t + o]; __syncthreads(); }
  float rms = rsqrtf(red[0] / (float)H + 1e-5f);
  bool vis = vmask ? (vmask[r] != 0) : true;
  __bf16 zero = f2bf(0.f);
  for (int i = t; i < H; i += 256) {
    __bf16 b = f2bf(row[i] * rms * w[i]);
    out_v[(size_t)r * H + i] = vis ? b : zero;
    if (out_l) out_l[(size_t)r * H + i] = vis ? zero : b;
  }
}

// ctx fp32 -> masked bf16 pair
__global__ void mask_convert(const float* __restrict__ x, const unsigned char* __restrict__ vmask,
                             __bf16* __restrict__ ov, __bf16* __restrict__ ol, int cols, size_t n) {
  __bf16 zero = f2bf(0.f);
  for (size_t i = (size_t)blockIdx.x * blockDim.x + threadIdx.x; i < n;
       i += (size_t)gridDim.x * blockDim.x) {
    size_t r = i / (size_t)cols;
    bool vis = vmask[r] != 0;
    __bf16 b = f2bf(x[i]);
    ov[i] = vis ? b : zero;
    ol[i] = vis ? zero : b;
  }
}

// mixed (S x 3H fp32) + positions -> rope'd q,k head-major [NH][S][HD] bf16,
// v transposed d-major [NH][HD][S] bf16 (so flash PV B-frags are contiguous).
__global__ void rope_pack(const float* __restrict__ mixed, const int* __restrict__ pos,
                          __bf16* __restrict__ qh, __bf16* __restrict__ kh, __bf16* __restrict__ vh) {
  int s = blockIdx.x, t = threadIdx.x;
  float p = (float)pos[s];
  const float* row = mixed + (size_t)s * (3 * H_);
  const float kln = 9.210340371976184f / 64.f;  // ln(10000)/half
  for (int idx = t; idx < H_ / 2; idx += 256) {
    int n = idx >> 6, d = idx & 63;
    float inv = __expf(-(float)d * kln);
    float fr = p * inv, sn, cs;
    __sincosf(fr, &sn, &cs);
    size_t ob = ((size_t)n * S_ + s) * HD_;
    float q1 = row[n * HD_ + d], q2 = row[n * HD_ + d + 64];
    qh[ob + d]      = f2bf(q1 * cs - q2 * sn);
    qh[ob + d + 64] = f2bf(q2 * cs + q1 * sn);
    float k1 = row[H_ + n * HD_ + d], k2 = row[H_ + n * HD_ + d + 64];
    kh[ob + d]      = f2bf(k1 * cs - k2 * sn);
    kh[ob + d + 64] = f2bf(k2 * cs + k1 * sn);
  }
  for (int idx = t; idx < H_; idx += 256) {
    int n = idx >> 7, d = idx & 127;
    vh[((size_t)n * HD_ + d) * S_ + s] = f2bf(row[2 * H_ + idx]);
  }
}

// fp32 (rows x sstride) token-major -> bf16 head-major [heads][rows][hd]
__global__ void split_heads(const float* __restrict__ src, __bf16* __restrict__ dst,
                            int rows, int heads, int hd, int sstride, int soff) {
  size_t n = (size_t)rows * heads * hd;
  for (size_t i = (size_t)blockIdx.x * blockDim.x + threadIdx.x; i < n;
       i += (size_t)gridDim.x * blockDim.x) {
    int d = (int)(i % hd);
    size_t tmp = i / hd;
    int h = (int)(tmp % heads);
    size_t r = tmp / heads;
    dst[((size_t)h * rows + r) * hd + d] = f2bf(src[r * (size_t)sstride + soff + h * hd + d]);
  }
}

// same but d-major output [heads][hd][rows] (for V)
__global__ void split_heads_t(const float* __restrict__ src, __bf16* __restrict__ dst,
                              int rows, int heads, int hd, int sstride, int soff) {
  size_t n = (size_t)rows * heads * hd;
  for (size_t i = (size_t)blockIdx.x * blockDim.x + threadIdx.x; i < n;
       i += (size_t)gridDim.x * blockDim.x) {
    int d = (int)(i % hd);
    size_t tmp = i / hd;
    int h = (int)(tmp % heads);
    size_t r = tmp / heads;
    dst[((size_t)h * hd + d) * rows + r] = f2bf(src[r * (size_t)sstride + soff + h * hd + d]);
  }
}

// gu (S x 2I fp32) -> silu(gate)*up bf16 (S x I)
__global__ void silu_mul(const float* __restrict__ gu, __bf16* __restrict__ act, int Iv) {
  size_t n = (size_t)S_ * Iv;
  for (size_t i = (size_t)blockIdx.x * blockDim.x + threadIdx.x; i < n;
       i += (size_t)gridDim.x * blockDim.x) {
    size_t s = i / (size_t)Iv;
    int c = (int)(i % (size_t)Iv);
    float g = gu[s * (size_t)(2 * Iv) + c];
    float u = gu[s * (size_t)(2 * Iv) + Iv + c];
    act[i] = f2bf(g * (1.f / (1.f + __expf(-g))) * u);
  }
}

// =====================================================================
// bf16 WMMA GEMM: C[MxN] = A0@W0 (+ A1@W1) (+ R), fp32 out, row-major.
// Weights pre-transposed: Wt[N][K] bf16, so B-fragments are contiguous
// 32B/lane global loads (L2-resident). A slab (64x32) double-buffered in
// LDS via async global->LDS copies. 256 thr = 8 waves, tile 64x128; each
// wave = 16 rows x 64 cols = 4 WMMA per K-chunk reusing one A-fragment.
// M%64==0, N%128==0, K%32==0.
// =====================================================================
#define LDT 40  // padded LDS stride in halfs (80B, 16B-aligned bases)

__global__ __launch_bounds__(256)
void gemm_dual(const __bf16* __restrict__ A0, const __bf16* __restrict__ Wt0,
               const __bf16* __restrict__ A1, const __bf16* __restrict__ Wt1,
               const float* __restrict__ R, float* __restrict__ C,
               int M, int N, int K) {
  __shared__ alignas(16) __bf16 As[2][64 * LDT];
  int t = threadIdx.x;
  int lane = t & 31, wv = t >> 5;
  int mi = wv & 3, nq = wv >> 2;
  int hi = (lane >> 4) & 1, l16 = lane & 15;
  int m0 = blockIdx.y * 64;
  int n0 = blockIdx.x * 128 + nq * 64;
  int arow = t >> 2, aseg = t & 3;  // staging: 16B per thread
  floatx8 acc[4];
  #pragma unroll
  for (int f = 0; f < 4; ++f) { floatx8 z = {}; acc[f] = z; }

  int CHN = K / 32;
  int total = (A1 ? 2 : 1) * CHN;
  // stage chunk 0
  stage16(A0 + (size_t)(m0 + arow) * K + aseg * 8, As[0] + arow * LDT + aseg * 8);
  stage_wait();
  __syncthreads();

  int cur = 0;
  for (int c = 0; c < total; ++c) {
    int pr = c / CHN, kc = (c % CHN) * 32;
    // prefetch next chunk into the other buffer (overlaps with WMMAs below)
    if (c + 1 < total) {
      int pn = (c + 1) / CHN, kn = ((c + 1) % CHN) * 32;
      const __bf16* An = pn ? A1 : A0;
      stage16(An + (size_t)(m0 + arow) * K + kn + aseg * 8,
              As[cur ^ 1] + arow * LDT + aseg * 8);
    }
    // A fragment (16x32) from LDS
    ABFrag a;
    const __bf16* ar = As[cur] + (mi * 16 + l16) * LDT;
    ((uintx4*)a.u)[0] = *(const uintx4*)(ar + (hi ? 16 : 0));
    ((uintx4*)a.u)[1] = *(const uintx4*)(ar + (hi ? 24 : 8));
    // 4 B fragments straight from global (transposed weights)
    const __bf16* Wt = pr ? Wt1 : Wt0;
    #pragma unroll
    for (int f = 0; f < 4; ++f) {
      ABFrag b;
      const __bf16* bp = Wt + (size_t)(n0 + f * 16 + l16) * K + kc + (hi ? 16 : 0);
      ((uintx4*)b.u)[0] = *(const uintx4*)(bp);
      ((uintx4*)b.u)[1] = *(const uintx4*)(bp + 8);
      acc[f] = wmma_bf16(a, b, acc[f]);
    }
    stage_wait();      // next-chunk async copy done
    __syncthreads();   // everyone done reading As[cur] / writing As[cur^1]
    cur ^= 1;
  }
  int rbase = m0 + mi * 16 + (hi ? 8 : 0);
  #pragma unroll
  for (int f = 0; f < 4; ++f) {
    int c0 = n0 + f * 16 + l16;
    #pragma unroll
    for (int i = 0; i < 8; ++i) {
      size_t off = (size_t)(rbase + i) * N + c0;
      float v = acc[f][i];
      if (R) v += R[off];
      C[off] = v;
    }
  }
}

// =====================================================================
// Flash attention, bf16 WMMA, fp32 online softmax. No inter-wave sync:
// each of the 4 waves owns 16 q-rows; P relayout goes through a private
// LDS slot (DS is in-order within a wave). K is key-major [h][kv][D];
// V is d-major [h][D][kv] so PV B-frags are contiguous global loads.
// =====================================================================
template <int D, bool CAUSAL>
__global__ __launch_bounds__(128)
void flash_attn(const __bf16* __restrict__ Q, const __bf16* __restrict__ Kp,
                const __bf16* __restrict__ Vt, float* __restrict__ Out,
                int Sq, int Skv, float scale, int ostride) {
  constexpr int DF = D / 16;   // output d-fragments per wave
  constexpr int QC = D / 32;   // K-dim chunks for QK^T
  __shared__ alignas(16) __bf16 pbuf[4 * 16 * 32]; // per-wave P tile
  int t = threadIdx.x, lane = t & 31, wv = t >> 5;
  int hi = (lane >> 4) & 1, l16 = lane & 15;
  int head = blockIdx.y;
  int q0 = blockIdx.x * 64 + wv * 16;

  // preload Q fragments (16 rows x D)
  ABFrag qa[QC];
  const __bf16* qrow = Q + ((size_t)head * Sq + q0 + l16) * D;
  #pragma unroll
  for (int c = 0; c < QC; ++c) {
    ((uintx4*)qa[c].u)[0] = *(const uintx4*)(qrow + c * 32 + (hi ? 16 : 0));
    ((uintx4*)qa[c].u)[1] = *(const uintx4*)(qrow + c * 32 + (hi ? 24 : 8));
  }
  floatx8 oacc[DF];
  #pragma unroll
  for (int dc = 0; dc < DF; ++dc) { floatx8 z = {}; oacc[dc] = z; }
  float m_run[8], l_run[8];
  #pragma unroll
  for (int i = 0; i < 8; ++i) { m_run[i] = -1e30f; l_run[i] = 0.f; }

  int kend = CAUSAL ? ((q0 + 16 + 31) & ~31) : Skv;  // per-wave causal bound
  __bf16* pw = pbuf + wv * 512;
  for (int kb = 0; kb < kend; kb += 32) {
    // scores: 16 q x 32 keys, two 16x16 C-fragments
    floatx8 s0 = {}, s1 = {};
    #pragma unroll
    for (int c = 0; c < QC; ++c) {
      ABFrag b0, b1;  // B = K^T: lane col = key, contiguous d run per lane
      const __bf16* kr0 = Kp + ((size_t)head * Skv + kb + l16) * D + c * 32 + (hi ? 16 : 0);
      ((uintx4*)b0.u)[0] = *(const uintx4*)(kr0);
      ((uintx4*)b0.u)[1] = *(const uintx4*)(kr0 + 8);
      const __bf16* kr1 = kr0 + 16 * D;
      ((uintx4*)b1.u)[0] = *(const uintx4*)(kr1);
      ((uintx4*)b1.u)[1] = *(const uintx4*)(kr1 + 8);
      s0 = wmma_bf16(qa[c], b0, s0);
      s1 = wmma_bf16(qa[c], b1, s1);
    }
    // online softmax per row (row i lives in one 16-lane half per ISA layout)
    int rb = q0 + (hi ? 8 : 0);
    #pragma unroll
    for (int i = 0; i < 8; ++i) {
      float a = s0[i] * scale, b = s1[i] * scale;
      if (CAUSAL) {
        int r = rb + i;
        if (kb + l16 > r)      a = -1e30f;
        if (kb + 16 + l16 > r) b = -1e30f;
      }
      float bm = fmaxf(a, b);
      #pragma unroll
      for (int o = 1; o < 16; o <<= 1) bm = fmaxf(bm, __shfl_xor(bm, o, 32));
      float mn = fmaxf(m_run[i], bm);
      float alpha = __expf(m_run[i] - mn);
      float pa = __expf(a - mn), pb = __expf(b - mn);
      float ls = pa + pb;
      #pragma unroll
      for (int o = 1; o < 16; o <<= 1) ls += __shfl_xor(ls, o, 32);
      l_run[i] = l_run[i] * alpha + ls;
      m_run[i] = mn;
      #pragma unroll
      for (int dc = 0; dc < DF; ++dc) oacc[dc][i] *= alpha;
      int m = i + (hi ? 8 : 0);
      pw[m * 32 + l16]      = f2bf(pa);
      pw[m * 32 + 16 + l16] = f2bf(pb);
    }
    // PV: P(16x32) @ V(32xD); P relayout via wave-private LDS (in-order DS)
    ABFrag pf;
    const __bf16* prr = pw + l16 * 32;
    ((uintx4*)pf.u)[0] = *(const uintx4*)(prr + (hi ? 16 : 0));
    ((uintx4*)pf.u)[1] = *(const uintx4*)(prr + (hi ? 24 : 8));
    #pragma unroll
    for (int dc = 0; dc < DF; ++dc) {
      ABFrag vb;  // d-major V: contiguous keys per lane
      const __bf16* vr = Vt + ((size_t)head * D + dc * 16 + l16) * Skv + kb + (hi ? 16 : 0);
      ((uintx4*)vb.u)[0] = *(const uintx4*)(vr);
      ((uintx4*)vb.u)[1] = *(const uintx4*)(vr + 8);
      oacc[dc] = wmma_bf16(pf, vb, oacc[dc]);
    }
  }
  // epilogue
  int rb = q0 + (hi ? 8 : 0);
  #pragma unroll
  for (int i = 0; i < 8; ++i) {
    float inv = 1.f / l_run[i];
    size_t ro = (size_t)(rb + i) * ostride + (size_t)head * D + l16;
    #pragma unroll
    for (int dc = 0; dc < DF; ++dc) Out[ro + dc * 16] = oacc[dc][i] * inv;
  }
}

// =====================================================================
// Host orchestration
// =====================================================================
extern "C" void kernel_launch(void* const* d_in, const int* in_sizes, int n_in,
                              void* d_out, int out_size, void* d_ws, size_t ws_size,
                              hipStream_t stream) {
  (void)in_sizes; (void)n_in; (void)out_size; (void)ws_size;
  const int*   positions = (const int*)d_in[0];
  const float* hs        = (const float*)d_in[1];
  const unsigned char* vmask = (const unsigned char*)d_in[2];  // numpy bool = 1B
  const float* enc  = (const float*)d_in[4];
  const float* wlni = (const float*)d_in[5];
  const float* wlpa = (const float*)d_in[6];
  const float* wlpc = (const float*)d_in[7];
  const float* wvqkv = (const float*)d_in[8];
  const float* wlqkv = (const float*)d_in[9];
  const float* wvd   = (const float*)d_in[10];
  const float* wld   = (const float*)d_in[11];
  const float* wcq   = (const float*)d_in[12];
  const float* wckv  = (const float*)d_in[13];
  const float* wcd   = (const float*)d_in[14];
  const float* wvgu  = (const float*)d_in[15];
  const float* wvdn  = (const float*)d_in[16];
  const float* wlgu  = (const float*)d_in[17];
  const float* wldn  = (const float*)d_in[18];
  float* out = (float*)d_out;

  uintptr_t bump = (uintptr_t)d_ws;
  auto alloc = [&](size_t bytes) -> void* {
    uintptr_t q = (bump + 255) & ~(uintptr_t)255;
    bump = q + bytes;
    return (void*)q;
  };
  auto abf  = [&](size_t e) -> __bf16* { return (__bf16*)alloc(e * 2); };
  auto af32 = [&](size_t e) -> float*  { return (float*)alloc(e * 4); };

  // bf16 transposed weights Wt[N][K]
  __bf16* bwvqkv = abf((size_t)H_ * 3 * H_);
  __bf16* bwlqkv = abf((size_t)H_ * 3 * H_);
  __bf16* bwvd   = abf((size_t)H_ * H_);
  __bf16* bwld   = abf((size_t)H_ * H_);
  __bf16* bwcq   = abf((size_t)H_ * CC_);
  __bf16* bwckv  = abf((size_t)CH_ * 2 * CC_);
  __bf16* bwcd   = abf((size_t)CC_ * H_);
  __bf16* bwvgu  = abf((size_t)H_ * 2 * I_);
  __bf16* bwlgu  = abf((size_t)H_ * 2 * I_);
  __bf16* bwvdn  = abf((size_t)I_ * H_);
  __bf16* bwldn  = abf((size_t)I_ * H_);
  __bf16* benc   = abf((size_t)E_ * CH_);
  // activations
  __bf16* h_v   = abf((size_t)S_ * H_);
  __bf16* h_l   = abf((size_t)S_ * H_);
  float*  mixed = af32((size_t)S_ * 3 * H_);
  __bf16* qh    = abf((size_t)S_ * H_);
  __bf16* kh    = abf((size_t)S_ * H_);
  __bf16* vh    = abf((size_t)S_ * H_);   // [NH][HD][S]
  float*  ctx   = af32((size_t)S_ * H_);
  __bf16* ctx_v = abf((size_t)S_ * H_);
  __bf16* ctx_l = abf((size_t)S_ * H_);
  float*  hid1  = af32((size_t)S_ * H_);
  __bf16* h2    = abf((size_t)S_ * H_);
  float*  cq    = af32((size_t)S_ * CC_);
  float*  kv    = af32((size_t)E_ * 2 * CC_);
  __bf16* cqh   = abf((size_t)S_ * CC_);
  __bf16* ckh   = abf((size_t)E_ * CC_);
  __bf16* cvh   = abf((size_t)E_ * CC_);  // [NH][CHD][E]
  float*  cctx  = af32((size_t)S_ * CC_);
  __bf16* cctxb = abf((size_t)S_ * CC_);
  float*  hid2  = af32((size_t)S_ * H_);
  __bf16* h3_v  = abf((size_t)S_ * H_);
  __bf16* h3_l  = abf((size_t)S_ * H_);
  float*  gu    = af32((size_t)S_ * 2 * I_);   // reused for both branches
  __bf16* act_v = abf((size_t)S_ * I_);
  __bf16* act_l = abf((size_t)S_ * I_);

  const int EL = 2048;
  dim3 tb(32, 8);
  auto tgrid = [](int K, int N) { return dim3(N / 32, K / 32); };
  // weight conversion + transpose; encoder plain conversion
  cvt_transpose<<<tgrid(H_, 3 * H_), tb, 0, stream>>>(wvqkv, bwvqkv, H_, 3 * H_);
  cvt_transpose<<<tgrid(H_, 3 * H_), tb, 0, stream>>>(wlqkv, bwlqkv, H_, 3 * H_);
  cvt_transpose<<<tgrid(H_, H_),     tb, 0, stream>>>(wvd,   bwvd,   H_, H_);
  cvt_transpose<<<tgrid(H_, H_),     tb, 0, stream>>>(wld,   bwld,   H_, H_);
  cvt_transpose<<<tgrid(H_, CC_),    tb, 0, stream>>>(wcq,   bwcq,   H_, CC_);
  cvt_transpose<<<tgrid(CH_, 2 * CC_), tb, 0, stream>>>(wckv, bwckv, CH_, 2 * CC_);
  cvt_transpose<<<tgrid(CC_, H_),    tb, 0, stream>>>(wcd,   bwcd,   CC_, H_);
  cvt_transpose<<<tgrid(H_, 2 * I_), tb, 0, stream>>>(wvgu,  bwvgu,  H_, 2 * I_);
  cvt_transpose<<<tgrid(H_, 2 * I_), tb, 0, stream>>>(wlgu,  bwlgu,  H_, 2 * I_);
  cvt_transpose<<<tgrid(I_, H_),     tb, 0, stream>>>(wvdn,  bwvdn,  I_, H_);
  cvt_transpose<<<tgrid(I_, H_),     tb, 0, stream>>>(wldn,  bwldn,  I_, H_);
  cvt_f32_bf16<<<EL, 256, 0, stream>>>(enc, benc, (size_t)E_ * CH_);

  // ---- self attention block ----
  rmsnorm_mask<<<S_, 256, 0, stream>>>(hs, wlni, vmask, h_v, h_l, H_);
  gemm_dual<<<dim3(3 * H_ / 128, S_ / 64), 256, 0, stream>>>(
      h_v, bwvqkv, h_l, bwlqkv, nullptr, mixed, S_, 3 * H_, H_);
  rope_pack<<<S_, 256, 0, stream>>>(mixed, positions, qh, kh, vh);
  flash_attn<HD_, true><<<dim3(S_ / 64, NH_), 128, 0, stream>>>(
      qh, kh, vh, ctx, S_, S_, 0.08838834764831845f /*1/sqrt(128)*/, H_);
  mask_convert<<<EL, 256, 0, stream>>>(ctx, vmask, ctx_v, ctx_l, H_, (size_t)S_ * H_);
  gemm_dual<<<dim3(H_ / 128, S_ / 64), 256, 0, stream>>>(
      ctx_v, bwvd, ctx_l, bwld, hs /*residual*/, hid1, S_, H_, H_);

  // ---- cross attention block ----
  rmsnorm_mask<<<S_, 256, 0, stream>>>(hid1, wlpa, nullptr, h2, nullptr, H_);
  gemm_dual<<<dim3(CC_ / 128, S_ / 64), 256, 0, stream>>>(
      h2, bwcq, nullptr, nullptr, nullptr, cq, S_, CC_, H_);
  gemm_dual<<<dim3(2 * CC_ / 128, E_ / 64), 256, 0, stream>>>(
      benc, bwckv, nullptr, nullptr, nullptr, kv, E_, 2 * CC_, CH_);
  split_heads<<<EL, 256, 0, stream>>>(cq, cqh, S_, NH_, CHD_, CC_,     0);
  split_heads<<<EL, 256, 0, stream>>>(kv, ckh, E_, NH_, CHD_, 2 * CC_, 0);
  split_heads_t<<<EL, 256, 0, stream>>>(kv, cvh, E_, NH_, CHD_, 2 * CC_, CC_);
  flash_attn<CHD_, false><<<dim3(S_ / 64, NH_), 128, 0, stream>>>(
      cqh, ckh, cvh, cctx, S_, E_, 0.125f /*1/sqrt(64)*/, CC_);
  cvt_f32_bf16<<<EL, 256, 0, stream>>>(cctx, cctxb, (size_t)S_ * CC_);
  gemm_dual<<<dim3(H_ / 128, S_ / 64), 256, 0, stream>>>(
      cctxb, bwcd, nullptr, nullptr, hid1 /*residual*/, hid2, S_, H_, CC_);

  // ---- dual MLP block ----
  rmsnorm_mask<<<S_, 256, 0, stream>>>(hid2, wlpc, vmask, h3_v, h3_l, H_);
  gemm_dual<<<dim3(2 * I_ / 128, S_ / 64), 256, 0, stream>>>(
      h3_v, bwvgu, nullptr, nullptr, nullptr, gu, S_, 2 * I_, H_);
  silu_mul<<<EL, 256, 0, stream>>>(gu, act_v, I_);
  gemm_dual<<<dim3(2 * I_ / 128, S_ / 64), 256, 0, stream>>>(
      h3_l, bwlgu, nullptr, nullptr, nullptr, gu, S_, 2 * I_, H_);
  silu_mul<<<EL, 256, 0, stream>>>(gu, act_l, I_);
  gemm_dual<<<dim3(H_ / 128, S_ / 64), 256, 0, stream>>>(
      act_v, bwvdn, act_l, bwldn, hid2 /*residual*/, out, S_, H_, I_);
}
